// SimpleRNN_22411139350579
// MI455X (gfx1250) — matrix operand order repaired and checked by reading, not compiled, not run on hardware
//
#include <hip/hip_runtime.h>
#include <hip/hip_bf16.h>

// Problem constants (from reference): T=256, B=128, V=79, H=1024
#define T_STEPS 256
#define BATCH   128
#define VEC     79
#define HID     1024
#define G4H     4096      // 4*H
#define KPAD    96        // V padded up to multiple of 32 for WMMA K

typedef __attribute__((ext_vector_type(16))) __bf16 bf16x16;
typedef __attribute__((ext_vector_type(8)))  float  f32x8;

union Frag16 { uint4 u[2]; bf16x16 v; };

// ---- WMMA fragment loaders (CDNA5 wave32 layouts, 16-bit) -----------------
// A matrix 16x32 (M x K), row-major source, ld in elements.
// lane<16:  M=lane,    elems 0..7 = K k0..k0+7,   elems 8..15 = K k0+16..k0+23
// lane>=16: M=lane&15, elems 0..7 = K k0+8..k0+15, elems 8..15 = K k0+24..k0+31
static __device__ inline bf16x16 load_frag_a(const __bf16* __restrict__ base,
                                             int ld, int row, int k0, int lane) {
    const int half = (lane >> 4) & 1;
    const __bf16* p = base + (long)row * ld + k0 + half * 8;
    Frag16 t;
    t.u[0] = *(const uint4*)(p);        // K base .. base+7
    t.u[1] = *(const uint4*)(p + 16);   // K base+16 .. base+23
    return t.v;
}

// B matrix 32x16 (K x N); column n of B is stored as row (ncol_base+n) of
// 'base' (i.e. B[k,n] = base[(ncol_base+n)*ld + k], contiguous in k).
// lane<16: N=lane, elems 0..7 = K k0..k0+7, 8..15 = K k0+16..k0+23 ; lane>=16: K k0+8..k0+15, k0+24..k0+31
static __device__ inline bf16x16 load_frag_b(const __bf16* __restrict__ base,
                                             int ld, int ncol_base, int k0, int lane) {
    const int n  = lane & 15;
    const int kh = (lane >> 4) << 3;
    const __bf16* p = base + (long)(ncol_base + n) * ld + k0 + kh;
    Frag16 t;
    t.u[0] = *(const uint4*)(p);        // K .. +7
    t.u[1] = *(const uint4*)(p + 16);   // K +16 .. +23
    return t.v;
}

static __device__ inline f32x8 wmma_bf16(bf16x16 a, bf16x16 b, f32x8 c) {
    return __builtin_amdgcn_wmma_f32_16x16x32_bf16(
        /*neg_a=*/false, a, /*neg_b=*/false, b,
        /*c_mod=*/(short)0, c, /*reuse_a=*/false, /*reuse_b=*/false);
}

static __device__ inline float sigmoidf_(float x) {
    return 1.0f / (1.0f + __expf(-x));
}

// ---------------------------------------------------------------------------
// Prep kernels
// ---------------------------------------------------------------------------
__global__ void zero_state_kernel(__bf16* hb0, float* c, float* h) {
    int i = blockIdx.x * blockDim.x + threadIdx.x;
    if (i < BATCH * HID) { hb0[i] = (__bf16)0.0f; c[i] = 0.0f; h[i] = 0.0f; }
}

__global__ void pack_inputs_kernel(const float* __restrict__ in, __bf16* __restrict__ out) {
    int i = blockIdx.x * blockDim.x + threadIdx.x;          // [T*B*KPAD]
    if (i >= T_STEPS * BATCH * KPAD) return;
    int k = i % KPAD, tb = i / KPAD;
    out[i] = (k < VEC) ? (__bf16)in[(long)tb * VEC + k] : (__bf16)0.0f;
}

__global__ void pack_whh_kernel(const float* __restrict__ w, __bf16* __restrict__ o) {
    int i = blockIdx.x * blockDim.x + threadIdx.x;          // [4H*H]
    if (i < G4H * HID) o[i] = (__bf16)w[i];
}

__global__ void pack_wout_kernel(const float* __restrict__ w, __bf16* __restrict__ o) {
    int i = blockIdx.x * blockDim.x + threadIdx.x;          // [80*H], row 79 zero
    if (i >= 80 * HID) return;
    int n = i / HID, k = i % HID;
    o[i] = (n < VEC) ? (__bf16)w[(long)n * HID + k] : (__bf16)0.0f;
}

// W_comb[g,v] = sum_h W_ih[g,h] * W_in[h,v]   (stored bf16, [4H, KPAD], v>=VEC zero)
__global__ void wcomb_kernel(const float* __restrict__ wih, const float* __restrict__ win,
                             __bf16* __restrict__ o) {
    int i = blockIdx.x * blockDim.x + threadIdx.x;          // [4H*KPAD]
    if (i >= G4H * KPAD) return;
    int v = i % KPAD, g = i / KPAD;
    float acc = 0.0f;
    if (v < VEC) {
        const float* wr = wih + (long)g * HID;
        for (int h = 0; h < HID; ++h) acc += wr[h] * win[(long)h * VEC + v];
    }
    o[i] = (__bf16)acc;
}

// b_comb[g] = sum_h W_ih[g,h]*b_in[h] + b_ih[g] + b_hh[g]
__global__ void bcomb_kernel(const float* __restrict__ wih, const float* __restrict__ bin,
                             const float* __restrict__ bih, const float* __restrict__ bhh,
                             float* __restrict__ o) {
    int g = blockIdx.x * blockDim.x + threadIdx.x;
    if (g >= G4H) return;
    float acc = bih[g] + bhh[g];
    const float* wr = wih + (long)g * HID;
    for (int h = 0; h < HID; ++h) acc += wr[h] * bin[h];
    o[g] = acc;
}

// ---------------------------------------------------------------------------
// Fused LSTM step: gates = x_t @ Wcomb^T + h @ Whh^T + bcomb ; cell update.
// Grid: 64 blocks; each block owns ALL 128 batch rows x 16 hidden cols x 4
// gates, so W_hh streams from L2 exactly once per step. Block = 8 waves
// (256 thr): wave = (gate, row-half); each wave computes a 64x16 tile.
// Fragments are explicitly double-buffered: loads for k+32 issue before the
// WMMAs of k consume their (already-landed) fragments.
// ---------------------------------------------------------------------------
__global__ void __launch_bounds__(256)
lstm_step_kernel(const __bf16* __restrict__ hprev,   // [B,H] bf16
                 __bf16*       __restrict__ hnext,   // [B,H] bf16
                 const __bf16* __restrict__ whh,     // [4H,H] bf16 (B cols contiguous in k)
                 const __bf16* __restrict__ xin,     // [B,KPAD] bf16 (this timestep)
                 const __bf16* __restrict__ wcomb,   // [4H,KPAD] bf16
                 const float*  __restrict__ bcomb,   // [4H]
                 float*        __restrict__ cstate,  // [B,H] f32
                 float*        __restrict__ hstate,  // [B,H] f32
                 __bf16*       __restrict__ hsout)   // [B,H] bf16 slice of hs[t], or null
{
    __shared__ float ldsAcc[4][BATCH][16];           // 32 KB

    const int lane = threadIdx.x & 31;
    const int wv   = threadIdx.x >> 5;               // 0..7
    const int gate = wv >> 1;                        // 0..3 (i,f,g,o)
    const int mh   = (wv & 1) * 64;                  // row-half base
    const int h0   = blockIdx.x * 16;                // hidden col tile base
    const int row  = lane & 15;
    const int nrow = gate * HID + h0;                // W_hh row base = gate cols

    f32x8 acc[4];
    #pragma unroll
    for (int m = 0; m < 4; ++m) acc[m] = (f32x8){0,0,0,0,0,0,0,0};

    // ---- Recurrent GEMM with software pipelining over K ----
    bf16x16 bC  = load_frag_b(whh, HID, nrow, 0, lane);
    bf16x16 a0C = load_frag_a(hprev, HID, mh +  0 + row, 0, lane);
    bf16x16 a1C = load_frag_a(hprev, HID, mh + 16 + row, 0, lane);
    bf16x16 a2C = load_frag_a(hprev, HID, mh + 32 + row, 0, lane);
    bf16x16 a3C = load_frag_a(hprev, HID, mh + 48 + row, 0, lane);
    for (int k0 = 0; k0 < HID; k0 += 32) {
        const int kn = (k0 + 32 < HID) ? (k0 + 32) : k0;   // clamp (redundant last reload)
        bf16x16 bN  = load_frag_b(whh, HID, nrow, kn, lane);
        bf16x16 a0N = load_frag_a(hprev, HID, mh +  0 + row, kn, lane);
        bf16x16 a1N = load_frag_a(hprev, HID, mh + 16 + row, kn, lane);
        bf16x16 a2N = load_frag_a(hprev, HID, mh + 32 + row, kn, lane);
        bf16x16 a3N = load_frag_a(hprev, HID, mh + 48 + row, kn, lane);
        acc[0] = wmma_bf16(a0C, bC, acc[0]);
        acc[1] = wmma_bf16(a1C, bC, acc[1]);
        acc[2] = wmma_bf16(a2C, bC, acc[2]);
        acc[3] = wmma_bf16(a3C, bC, acc[3]);
        bC = bN; a0C = a0N; a1C = a1N; a2C = a2N; a3C = a3N;
    }

    // ---- Input GEMM (folded W_ih*W_in), same pipelining, K = 96 ----
    bC  = load_frag_b(wcomb, KPAD, nrow, 0, lane);
    a0C = load_frag_a(xin, KPAD, mh +  0 + row, 0, lane);
    a1C = load_frag_a(xin, KPAD, mh + 16 + row, 0, lane);
    a2C = load_frag_a(xin, KPAD, mh + 32 + row, 0, lane);
    a3C = load_frag_a(xin, KPAD, mh + 48 + row, 0, lane);
    for (int k0 = 0; k0 < KPAD; k0 += 32) {
        const int kn = (k0 + 32 < KPAD) ? (k0 + 32) : k0;
        bf16x16 bN  = load_frag_b(wcomb, KPAD, nrow, kn, lane);
        bf16x16 a0N = load_frag_a(xin, KPAD, mh +  0 + row, kn, lane);
        bf16x16 a1N = load_frag_a(xin, KPAD, mh + 16 + row, kn, lane);
        bf16x16 a2N = load_frag_a(xin, KPAD, mh + 32 + row, kn, lane);
        bf16x16 a3N = load_frag_a(xin, KPAD, mh + 48 + row, kn, lane);
        acc[0] = wmma_bf16(a0C, bC, acc[0]);
        acc[1] = wmma_bf16(a1C, bC, acc[1]);
        acc[2] = wmma_bf16(a2C, bC, acc[2]);
        acc[3] = wmma_bf16(a3C, bC, acc[3]);
        bC = bN; a0C = a0N; a1C = a1N; a2C = a2N; a3C = a3N;
    }

    // Stage accumulators: VGPR r, lanes 0-15 -> (M=r, N=lane); 16-31 -> (M=r+8)
    const int mofs = (lane >> 4) * 8;
    #pragma unroll
    for (int m = 0; m < 4; ++m)
        #pragma unroll
        for (int r = 0; r < 8; ++r)
            ldsAcc[gate][mh + m * 16 + r + mofs][row] = acc[m][r];
    __syncthreads();

    // LSTM cell on the 128x16 tile (torch gate order i,f,g,o)
    for (int e = threadIdx.x; e < BATCH * 16; e += 256) {
        int m = e >> 4, n = e & 15;
        int hc = h0 + n;
        float gi = ldsAcc[0][m][n] + bcomb[hc];
        float gf = ldsAcc[1][m][n] + bcomb[HID + hc];
        float gg = ldsAcc[2][m][n] + bcomb[2 * HID + hc];
        float go = ldsAcc[3][m][n] + bcomb[3 * HID + hc];
        float iv = sigmoidf_(gi);
        float fv = sigmoidf_(gf);
        float gv = tanhf(gg);
        float ov = sigmoidf_(go);
        int idx = m * HID + hc;
        float cn = fv * cstate[idx] + iv * gv;
        float hn = ov * tanhf(cn);
        cstate[idx] = cn;
        hstate[idx] = hn;
        __bf16 hb = (__bf16)hn;
        hnext[idx] = hb;
        if (hsout) hsout[idx] = hb;
    }
}

// ---------------------------------------------------------------------------
// Batched output projection over ALL timesteps: outputs = hs @ W_out^T + b_out
// M = T*B = 32768 (2048 M-tiles), N = 79 (padded 80 -> 5 subtiles), K = 1024.
// Grid: 512 blocks x 4 waves; wave handles one 16x80 tile. Same pipelining.
// ---------------------------------------------------------------------------
__global__ void __launch_bounds__(128)
out_gemm_all_kernel(const __bf16* __restrict__ hs,     // [T*B, H] bf16
                    const __bf16* __restrict__ wout,   // [80,H] bf16 (row 79 zero)
                    const float*  __restrict__ bout,   // [79]
                    float*        __restrict__ out)    // [T*B, VEC]
{
    const int lane = threadIdx.x & 31;
    const int wave = threadIdx.x >> 5;
    const int m0   = (blockIdx.x * 4 + wave) * 16;
    const int row  = lane & 15;

    f32x8 acc[5];
    #pragma unroll
    for (int s = 0; s < 5; ++s) acc[s] = (f32x8){0,0,0,0,0,0,0,0};

    bf16x16 aC  = load_frag_a(hs, HID, m0 + row, 0, lane);
    bf16x16 b0C = load_frag_b(wout, HID,  0, 0, lane);
    bf16x16 b1C = load_frag_b(wout, HID, 16, 0, lane);
    bf16x16 b2C = load_frag_b(wout, HID, 32, 0, lane);
    bf16x16 b3C = load_frag_b(wout, HID, 48, 0, lane);
    bf16x16 b4C = load_frag_b(wout, HID, 64, 0, lane);
    for (int k0 = 0; k0 < HID; k0 += 32) {
        const int kn = (k0 + 32 < HID) ? (k0 + 32) : k0;
        bf16x16 aN  = load_frag_a(hs, HID, m0 + row, kn, lane);
        bf16x16 b0N = load_frag_b(wout, HID,  0, kn, lane);
        bf16x16 b1N = load_frag_b(wout, HID, 16, kn, lane);
        bf16x16 b2N = load_frag_b(wout, HID, 32, kn, lane);
        bf16x16 b3N = load_frag_b(wout, HID, 48, kn, lane);
        bf16x16 b4N = load_frag_b(wout, HID, 64, kn, lane);
        acc[0] = wmma_bf16(aC, b0C, acc[0]);
        acc[1] = wmma_bf16(aC, b1C, acc[1]);
        acc[2] = wmma_bf16(aC, b2C, acc[2]);
        acc[3] = wmma_bf16(aC, b3C, acc[3]);
        acc[4] = wmma_bf16(aC, b4C, acc[4]);
        aC = aN; b0C = b0N; b1C = b1N; b2C = b2N; b3C = b3N; b4C = b4N;
    }

    const int mofs = (lane >> 4) * 8;
    #pragma unroll
    for (int s = 0; s < 5; ++s) {
        int n = s * 16 + row;
        if (n < VEC) {
            float bb = bout[n];
            #pragma unroll
            for (int r = 0; r < 8; ++r)
                out[(long)(m0 + r + mofs) * VEC + n] = acc[s][r] + bb;
        }
    }
}

// ---------------------------------------------------------------------------
// Fallback per-step output projection (used only when ws is too small for hs)
// ---------------------------------------------------------------------------
__global__ void __launch_bounds__(32)
out_proj_kernel(const __bf16* __restrict__ hcur,    // [B,H] bf16
                const __bf16* __restrict__ wout,    // [80,H] bf16 (row 79 zero)
                const float*  __restrict__ bout,    // [79]
                float*        __restrict__ out)     // [B,VEC] (this timestep)
{
    const int lane = threadIdx.x & 31;
    const int m0   = blockIdx.x * 16;
    const int row  = lane & 15;

    f32x8 acc[5];
    #pragma unroll
    for (int s = 0; s < 5; ++s) acc[s] = (f32x8){0,0,0,0,0,0,0,0};

    bf16x16 aC  = load_frag_a(hcur, HID, m0 + row, 0, lane);
    bf16x16 b0C = load_frag_b(wout, HID,  0, 0, lane);
    bf16x16 b1C = load_frag_b(wout, HID, 16, 0, lane);
    bf16x16 b2C = load_frag_b(wout, HID, 32, 0, lane);
    bf16x16 b3C = load_frag_b(wout, HID, 48, 0, lane);
    bf16x16 b4C = load_frag_b(wout, HID, 64, 0, lane);
    for (int k0 = 0; k0 < HID; k0 += 32) {
        const int kn = (k0 + 32 < HID) ? (k0 + 32) : k0;
        bf16x16 aN  = load_frag_a(hcur, HID, m0 + row, kn, lane);
        bf16x16 b0N = load_frag_b(wout, HID,  0, kn, lane);
        bf16x16 b1N = load_frag_b(wout, HID, 16, kn, lane);
        bf16x16 b2N = load_frag_b(wout, HID, 32, kn, lane);
        bf16x16 b3N = load_frag_b(wout, HID, 48, kn, lane);
        bf16x16 b4N = load_frag_b(wout, HID, 64, kn, lane);
        acc[0] = wmma_bf16(aC, b0C, acc[0]);
        acc[1] = wmma_bf16(aC, b1C, acc[1]);
        acc[2] = wmma_bf16(aC, b2C, acc[2]);
        acc[3] = wmma_bf16(aC, b3C, acc[3]);
        acc[4] = wmma_bf16(aC, b4C, acc[4]);
        aC = aN; b0C = b0N; b1C = b1N; b2C = b2N; b3C = b3N; b4C = b4N;
    }

    const int mofs = (lane >> 4) * 8;
    #pragma unroll
    for (int s = 0; s < 5; ++s) {
        int n = s * 16 + row;
        if (n < VEC) {
            float bb = bout[n];
            #pragma unroll
            for (int r = 0; r < 8; ++r)
                out[(m0 + r + mofs) * VEC + n] = acc[s][r] + bb;
        }
    }
}

__global__ void finalize_kernel(const float* __restrict__ hstate,
                                const float* __restrict__ cstate,
                                float* __restrict__ out_tail) {
    int i = blockIdx.x * blockDim.x + threadIdx.x;
    if (i < BATCH * HID) {
        out_tail[i]               = hstate[i];   // hx
        out_tail[BATCH * HID + i] = cstate[i];   // cx
    }
}

// ---------------------------------------------------------------------------
// Launch
// ---------------------------------------------------------------------------
extern "C" void kernel_launch(void* const* d_in, const int* in_sizes, int n_in,
                              void* d_out, int out_size, void* d_ws, size_t ws_size,
                              hipStream_t stream) {
    const float* inputs = (const float*)d_in[0];   // [T,B,V]
    const float* W_in   = (const float*)d_in[1];   // [H,V]
    const float* b_in   = (const float*)d_in[2];   // [H]
    const float* W_ih   = (const float*)d_in[3];   // [4H,H]
    const float* b_ih   = (const float*)d_in[4];   // [4H]
    const float* W_hh   = (const float*)d_in[5];   // [4H,H]
    const float* b_hh   = (const float*)d_in[6];   // [4H]
    const float* W_out  = (const float*)d_in[7];   // [V,H]
    const float* b_out  = (const float*)d_in[8];   // [V]
    float* out = (float*)d_out;                    // outputs | hx | cx

    // Workspace carve-up (all offsets 16B aligned)
    char* ws = (char*)d_ws;
    __bf16* hb0     = (__bf16*)(ws + 0);                  // 262144 B
    __bf16* hb1     = (__bf16*)(ws + 262144);             // 262144 B
    float*  cstate  = (float*) (ws + 524288);             // 524288 B
    float*  hstate  = (float*) (ws + 1048576);            // 524288 B
    __bf16* in_bf16 = (__bf16*)(ws + 1572864);            // 6291456 B
    __bf16* wcomb   = (__bf16*)(ws + 7864320);            // 786432 B
    float*  bcomb   = (float*) (ws + 8650752);            // 16384 B
    __bf16* whh16   = (__bf16*)(ws + 8667136);            // 8388608 B
    __bf16* wout16  = (__bf16*)(ws + 17055744);           // 163840 B
    __bf16* hs      = (__bf16*)(ws + 17219584);           // optional: 67108864 B
    const size_t FULL_NEED = 17219584ull + (size_t)T_STEPS * BATCH * HID * 2;
    const bool batched = (ws_size >= FULL_NEED);          // fixed per process -> deterministic

    // ---- prep ----
    zero_state_kernel<<<(BATCH * HID + 255) / 256, 256, 0, stream>>>(hb0, cstate, hstate);
    pack_inputs_kernel<<<(T_STEPS * BATCH * KPAD + 255) / 256, 256, 0, stream>>>(inputs, in_bf16);
    pack_whh_kernel<<<(G4H * HID + 255) / 256, 256, 0, stream>>>(W_hh, whh16);
    pack_wout_kernel<<<(80 * HID + 255) / 256, 256, 0, stream>>>(W_out, wout16);
    wcomb_kernel<<<(G4H * KPAD + 255) / 256, 256, 0, stream>>>(W_ih, W_in, wcomb);
    bcomb_kernel<<<(G4H + 255) / 256, 256, 0, stream>>>(W_ih, b_in, b_ih, b_hh, bcomb);

    // ---- recurrent scan ----
    for (int t = 0; t < T_STEPS; ++t) {
        __bf16* hprev = (t & 1) ? hb1 : hb0;
        __bf16* hnext = (t & 1) ? hb0 : hb1;
        __bf16* hsout = batched ? (hs + (long)t * BATCH * HID) : (__bf16*)nullptr;
        lstm_step_kernel<<<HID / 16, 256, 0, stream>>>(
            hprev, hnext, whh16,
            in_bf16 + (long)t * BATCH * KPAD, wcomb, bcomb,
            cstate, hstate, hsout);
        if (!batched) {
            out_proj_kernel<<<BATCH / 16, 32, 0, stream>>>(
                hnext, wout16, b_out, out + (long)t * BATCH * VEC);
        }
    }

    // ---- output projection (batched path): 32768 x 80 x 1024 GEMM ----
    if (batched) {
        out_gemm_all_kernel<<<(T_STEPS * BATCH) / 64, 128, 0, stream>>>(
            hs, wout16, b_out, out);
    }

    // ---- hx, cx ----
    finalize_kernel<<<(BATCH * HID + 255) / 256, 256, 0, stream>>>(
        hstate, cstate, out + (long)T_STEPS * BATCH * VEC);
}
